// VectorQuantizer_6708738916533
// MI455X (gfx1250) — matrix operand order repaired and checked by reading, not compiled, not run on hardware
//
#include <hip/hip_runtime.h>

typedef __attribute__((ext_vector_type(16))) _Float16 v16h;
typedef __attribute__((ext_vector_type(8)))  _Float16 v8h;
typedef __attribute__((ext_vector_type(8)))  float    v8f;

#define CDIM   256
#define SDIM   16384            // 16*32*32
#define NROWS  131072           // 8*SDIM
#define KCODES 1024
#define MTILE  128
#define KCHUNK 128              // codes staged in LDS per outer iteration
#define BCS    33554432         // 8*256*16384
#define ENC_ELEMS 134217728ll   // NROWS*KCODES

// ---------------- init: zero counts + loss accumulator ----------------
__global__ void vq_init(float* __restrict__ counts, double* __restrict__ lossAcc) {
    int t = blockIdx.x * blockDim.x + threadIdx.x;
    if (t < KCODES) counts[t] = 0.0f;
    if (t == 0) *lossAcc = 0.0;
}

// ---------------- prep: codebook -> f16, per-code squared norms ----------------
__global__ __launch_bounds__(256) void vq_prep(const float* __restrict__ emb,
                                               _Float16* __restrict__ ebf16,
                                               float* __restrict__ enorm) {
    const int k = blockIdx.x;
    const int c = threadIdx.x;
    float v = emb[(size_t)k * CDIM + c];
    ebf16[(size_t)k * CDIM + c] = (_Float16)v;
    float sq = v * v;
    #pragma unroll
    for (int m = 16; m >= 1; m >>= 1) sq += __shfl_xor(sq, m, 32);
    __shared__ float wsum[8];
    const int wave = threadIdx.x >> 5, lane = threadIdx.x & 31;
    if (lane == 0) wsum[wave] = sq;
    __syncthreads();
    if (threadIdx.x == 0) {
        float s = 0.0f;
        for (int i = 0; i < 8; ++i) s += wsum[i];
        enorm[k] = s;
    }
}

// load the 8 B fragments of one 16-code k-tile from the LDS codebook chunk
__device__ __forceinline__ void loadBfrag(v16h* dst, const _Float16* tileB,
                                          int kt, int hf, int lm) {
    #pragma unroll
    for (int cc = 0; cc < 8; ++cc)
        dst[cc] = *(const v16h*)(tileB + (kt * 16 + lm) * CDIM + cc * 32 + hf * 16);
}

// ---------------- distance GEMM + argmin (WMMA f16 -> f32) ----------------
__global__ __launch_bounds__(256) void vq_dist(const float* __restrict__ x,
                                               const _Float16* __restrict__ ebf16,
                                               const float* __restrict__ enorm,
                                               int* __restrict__ indices) {
    __shared__ _Float16 tileA[MTILE * CDIM]  __attribute__((aligned(32)));  // 64 KB
    __shared__ _Float16 tileB[KCHUNK * CDIM] __attribute__((aligned(32)));  // 64 KB
    __shared__ float    tileEn[KCHUNK];

    const int row0 = blockIdx.x * MTILE;
    const int b    = row0 / SDIM;
    const int s0   = row0 % SDIM;                 // MTILE divides SDIM -> one b per tile
    const int wave = threadIdx.x >> 5;
    const int lane = threadIdx.x & 31;

    // stage input tile: coalesced float4 loads along spatial dim, f16 store [m][c]
    for (int cc = 0; cc < CDIM; cc += 8) {
        const int c = cc + wave;
        const float4* src = (const float4*)(x + ((size_t)(b * CDIM + c)) * SDIM + s0);
        float4 v = src[lane];
        const int m = lane * 4;
        tileA[(m + 0) * CDIM + c] = (_Float16)v.x;
        tileA[(m + 1) * CDIM + c] = (_Float16)v.y;
        tileA[(m + 2) * CDIM + c] = (_Float16)v.z;
        tileA[(m + 3) * CDIM + c] = (_Float16)v.w;
    }
    __syncthreads();

    const int m0 = wave * 16;
    const int hf = lane >> 4;        // half-wave select
    const int lm = lane & 15;        // column / row-in-tile

    // A fragments (16x32 f16 per C-chunk), resident across the whole K loop
    v16h afrag[8];
    #pragma unroll
    for (int cc = 0; cc < 8; ++cc) {
        const _Float16* pa = &tileA[(m0 + lm) * CDIM + cc * 32 + hf * 16];
        v8h lo = *(const v8h*)(pa);          // K = hf*16 .. +7
        v8h hi = *(const v8h*)(pa + 8);      // K = hf*16+8 .. +15
        afrag[cc] = __builtin_shufflevector(lo, hi, 0,1,2,3,4,5,6,7,
                                                    8,9,10,11,12,13,14,15);
    }

    float best[8];
    int   bidx[8];
    #pragma unroll
    for (int j = 0; j < 8; ++j) { best[j] = 3.0e38f; bidx[j] = 0; }

    for (int ch = 0; ch < KCODES / KCHUNK; ++ch) {
        __syncthreads();   // all waves done reading previous tileB

        // stage one codebook chunk (KCHUNK codes) into LDS, shared by all 8 waves
        {
            const float4* src = (const float4*)(ebf16 + (size_t)ch * KCHUNK * CDIM);
            float4*       dst = (float4*)tileB;
            #pragma unroll 4
            for (int i = threadIdx.x; i < KCHUNK * CDIM / 8; i += 256)
                dst[i] = src[i];
            if (threadIdx.x < KCHUNK)
                tileEn[threadIdx.x] = enorm[ch * KCHUNK + threadIdx.x];
        }
        // warm L2 for the next chunk while we compute this one
        if (ch + 1 < KCODES / KCHUNK) {
            const char* nxt = (const char*)(ebf16 + (size_t)(ch + 1) * KCHUNK * CDIM);
            __builtin_prefetch(nxt + threadIdx.x * 256, 0, 0);
        }
        __syncthreads();

        // ping-pong B fragments: fetch k-tile kt+1 from LDS while WMMAs run on kt
        v16h bfr[2][8];
        loadBfrag(bfr[0], tileB, 0, hf, lm);
        #pragma unroll
        for (int kt = 0; kt < KCHUNK / 16; ++kt) {
            if (kt + 1 < KCHUNK / 16)
                loadBfrag(bfr[(kt + 1) & 1], tileB, kt + 1, hf, lm);
            v8f acc = {};
            #pragma unroll
            for (int cc = 0; cc < 8; ++cc)
                acc = __builtin_amdgcn_wmma_f32_16x16x32_f16(
                          false, afrag[cc], false, bfr[kt & 1][cc],
                          (short)0, acc, false, false);
            const int   codeL = kt * 16 + lm;
            const float en    = tileEn[codeL];
            const int   code  = ch * KCHUNK + codeL;
            #pragma unroll
            for (int j = 0; j < 8; ++j) {
                float d = en - 2.0f * acc[j];    // ||x||^2 constant per row: drop it
                if (d < best[j]) { best[j] = d; bidx[j] = code; }
            }
        }
    }

    // min-reduce across the 16 columns of each half-wave
    #pragma unroll
    for (int mask = 1; mask <= 8; mask <<= 1) {
        #pragma unroll
        for (int j = 0; j < 8; ++j) {
            float ov = __shfl_xor(best[j], mask, 32);
            int   oi = __shfl_xor(bidx[j], mask, 32);
            if (ov < best[j] || (ov == best[j] && oi < bidx[j])) {
                best[j] = ov; bidx[j] = oi;
            }
        }
    }
    if (lm == 0) {
        #pragma unroll
        for (int j = 0; j < 8; ++j)
            indices[row0 + m0 + hf * 8 + j] = bidx[j];   // rows m0 + hf*8 + j
    }
}

// ---------------- gather codebook -> quantized output + loss accum ----------------
__global__ __launch_bounds__(256) void vq_quant(const float* __restrict__ x,
                                                const float* __restrict__ emb,
                                                const int* __restrict__ indices,
                                                float* __restrict__ qout,
                                                double* __restrict__ lossAcc) {
    const size_t tid = (size_t)blockIdx.x * 256 + threadIdx.x;
    const int s  = (int)(tid % SDIM);
    const int bc = (int)(tid / SDIM);
    const int c  = bc % CDIM;
    const int b  = bc / CDIM;
    const int code = indices[b * SDIM + s];
    const float qv = emb[(size_t)code * CDIM + c];
    const float xv = x[tid];
    qout[tid] = qv;                     // straight-through value == quantized
    float d  = qv - xv;
    float d2 = d * d;
    #pragma unroll
    for (int m = 16; m >= 1; m >>= 1) d2 += __shfl_xor(d2, m, 32);
    __shared__ float wsum[8];
    const int wave = threadIdx.x >> 5, lane = threadIdx.x & 31;
    if (lane == 0) wsum[wave] = d2;
    __syncthreads();
    if (threadIdx.x == 0) {
        float s8 = 0.0f;
        for (int i = 0; i < 8; ++i) s8 += wsum[i];
        atomicAdd(lossAcc, (double)s8);
    }
}

// ---------------- one-hot encodings: zero-fill + scatter ----------------
__global__ void vq_zero_enc(float2* __restrict__ enc2) {
    size_t t = (size_t)blockIdx.x * blockDim.x + threadIdx.x;
    enc2[t] = make_float2(0.0f, 0.0f);
}

__global__ void vq_scatter(const int* __restrict__ indices,
                           float* __restrict__ enc,
                           float* __restrict__ counts) {
    const int n = blockIdx.x * blockDim.x + threadIdx.x;
    if (n < NROWS) {
        const int code = indices[n];
        enc[(size_t)n * KCODES + code] = 1.0f;
        atomicAdd(&counts[code], 1.0f);   // integral sums: exact in f32
    }
}

// ---------------- loss + perplexity ----------------
__global__ __launch_bounds__(1024) void vq_finalize(const float* __restrict__ counts,
                                                    const double* __restrict__ lossAcc,
                                                    float* __restrict__ out_loss,
                                                    float* __restrict__ out_perp) {
    const int t = threadIdx.x;
    const float p = counts[t] * (1.0f / (float)NROWS);
    float term = p * logf(p + 1e-10f);
    #pragma unroll
    for (int m = 16; m >= 1; m >>= 1) term += __shfl_xor(term, m, 32);
    __shared__ float wsum[32];
    const int wave = t >> 5, lane = t & 31;
    if (lane == 0) wsum[wave] = term;
    __syncthreads();
    if (t == 0) {
        float s = 0.0f;
        for (int i = 0; i < 32; ++i) s += wsum[i];
        *out_perp = expf(-s);
        // q_latent_loss == e_latent_loss numerically -> loss = 1.25 * mean((q-x)^2)
        *out_loss = (float)(*lossAcc * (1.25 / (double)BCS));
    }
}

extern "C" void kernel_launch(void* const* d_in, const int* in_sizes, int n_in,
                              void* d_out, int out_size, void* d_ws, size_t ws_size,
                              hipStream_t stream) {
    (void)in_sizes; (void)n_in; (void)out_size; (void)ws_size;
    const float* x   = (const float*)d_in[0];   // [8,256,16,32,32]
    const float* emb = (const float*)d_in[1];   // [1024,256]
    float* out = (float*)d_out;

    // workspace: [enorm 4KB][counts 4KB][lossAcc 8B pad][indices 512KB][ebf16 512KB]
    char* ws = (char*)d_ws;
    float*    enorm   = (float*)(ws + 0);
    float*    counts  = (float*)(ws + 4096);
    double*   lossAcc = (double*)(ws + 8192);
    int*      indices = (int*)(ws + 12288);
    _Float16* ebf16   = (_Float16*)(ws + 12288 + (size_t)NROWS * 4);

    float* out_loss = out;                        // [0]
    float* qout     = out + 1;                    // [1 .. 1+BCS)
    float* out_perp = out + 1 + (size_t)BCS;      // [1+BCS]
    float* enc      = out + 2 + (size_t)BCS;      // [2+BCS ..) — 8B-aligned base

    vq_init    <<<(KCODES + 255) / 256, 256, 0, stream>>>(counts, lossAcc);
    vq_prep    <<<KCODES,             256, 0, stream>>>(emb, ebf16, enorm);
    vq_dist    <<<NROWS / MTILE,      256, 0, stream>>>(x, ebf16, enorm, indices);
    vq_quant   <<<BCS / 256,          256, 0, stream>>>(x, emb, indices, qout, lossAcc);
    vq_zero_enc<<<(unsigned)((ENC_ELEMS / 2) / 256), 256, 0, stream>>>((float2*)enc);
    vq_scatter <<<(NROWS + 255) / 256, 256, 0, stream>>>(indices, enc, counts);
    vq_finalize<<<1, 1024, 0, stream>>>(counts, lossAcc, out_loss, out_perp);
}
